// NKF_57844619543054
// MI455X (gfx1250) — compile-verified, matrix operand order checked by reading, not run on hardware
//
#include <hip/hip_runtime.h>
#include <hip/hip_bf16.h>

// ---------------------------------------------------------------------------
// Problem constants (from reference): B=32, T=512, N=512, K=32, H=64, D=2,
// PROJ=16, HALF=256.
// Input pointer order assumes insertion-order flattening of setup_inputs():
//   0:y 1:x  2..25: flow[l]{W1,b1,W2,b2,W3,b3} l=0..3
//   26:proj_W 27:proj_b
//   28..35: lstm[l]{Wih,bih,Whh,bhh} l=0..1
//   36:head_W 37:head_b
// ---------------------------------------------------------------------------

typedef __attribute__((ext_vector_type(16))) _Float16 v16h;
typedef __attribute__((ext_vector_type(8)))  float    v8f;

#define WMMA_F16(a, b, c) \
  __builtin_amdgcn_wmma_f32_16x16x32_f16(false, (a), false, (b), (short)0, (c), false, false)

constexpr int Bc = 32, Tc = 512, Nc = 512, Kc = 32, Hc = 64, PROJc = 16, HALFc = 256;
constexpr int BT = Bc * Tc;   // 16384 rows

// --- CDNA5 async global->LDS copy (ASYNCcnt path, ISA 08 §4) ----------------
// GV mode: vdst = per-lane LDS byte address, vaddr = 64-bit global address.
// Generic pointers to __shared__ map to LDS as addr[31:0] (ISA §10.2).
__device__ __forceinline__ void async_copy_b128(void* lds_dst, const void* gsrc) {
  uint32_t l = (uint32_t)(uintptr_t)lds_dst;
  uint64_t g = (uint64_t)(uintptr_t)gsrc;
  asm volatile("global_load_async_to_lds_b128 %0, %1, off"
               :: "v"(l), "v"(g) : "memory");
}
__device__ __forceinline__ void async_wait0() {
  asm volatile("s_wait_asynccnt 0x0" ::: "memory");
}

// --- fragment loaders (CDNA5 wave32 16x16x32 f16 layout) --------------------
// A/B 16x32 f16: lanes 0-15 -> row (lane&15), K 0..7 (halves 0..7) and 16..23
// (halves 8..15); lanes 16-31 -> same row, K 8..15 and 24..31.
__device__ __forceinline__ v16h load_frag_f16(const _Float16* p, int ld, int lane) {
  int r  = lane & 15;
  int k0 = (lane < 16) ? 0 : 8;
  const _Float16* q = p + (size_t)r * ld;
  v16h f;
#pragma unroll
  for (int i = 0; i < 8; ++i) f[i] = q[k0 + i];
#pragma unroll
  for (int i = 0; i < 8; ++i) f[8 + i] = q[k0 + 16 + i];
  return f;
}

__device__ __forceinline__ v16h load_frag_f32(const float* p, int ld, int lane) {
  int r  = lane & 15;
  int k0 = (lane < 16) ? 0 : 8;
  const float* q = p + (size_t)r * ld;
  v16h f;
#pragma unroll
  for (int i = 0; i < 8; ++i) f[i] = (_Float16)q[k0 + i];
#pragma unroll
  for (int i = 0; i < 8; ++i) f[8 + i] = (_Float16)q[k0 + 16 + i];
  return f;
}

// A-frag from the flow input (B*T, N) with optional channel reversal.
__device__ __forceinline__ v16h load_frag_y(const float* y, size_t row0, int col0,
                                            int permute, int lane) {
  int m  = lane & 15;
  int k0 = (lane < 16) ? 0 : 8;
  const float* r = y + (row0 + m) * (size_t)Nc;
  v16h f;
#pragma unroll
  for (int i = 0; i < 8; ++i) {
    int c = col0 + k0 + i;
    f[i] = (_Float16)r[permute ? (Nc - 1 - c) : c];
  }
#pragma unroll
  for (int i = 0; i < 8; ++i) {
    int c = col0 + k0 + 16 + i;
    f[8 + i] = (_Float16)r[permute ? (Nc - 1 - c) : c];
  }
  return f;
}

// --- one-time weight prep ---------------------------------------------------
__global__ void cvt_f16_kernel(const float* __restrict__ src, _Float16* __restrict__ dst, int n) {
  int i = blockIdx.x * blockDim.x + threadIdx.x;
  if (i < n) dst[i] = (_Float16)src[i];
}

// Fuse [Wih | Whh | zero-pad] -> (4H, kdim) f16 so the LSTM gate GEMM is one
// WMMA K-loop over the concatenated [x_t | h_{t-1}] operand.
__global__ void pack_lstm_kernel(const float* __restrict__ Wih, const float* __restrict__ Whh,
                                 _Float16* __restrict__ dst, int in_dim, int kdim) {
  int i = blockIdx.x * blockDim.x + threadIdx.x;
  if (i >= 4 * Hc * kdim) return;
  int o = i / kdim, k = i % kdim;
  float v = 0.f;
  if (k < in_dim)           v = Wih[(size_t)o * in_dim + k];
  else if (k < in_dim + Hc) v = Whh[(size_t)o * Hc + (k - in_dim)];
  dst[i] = (_Float16)v;
}

__global__ void zero_kernel(float* p) {
  if (blockIdx.x == 0 && threadIdx.x == 0) p[0] = 0.f;
}

// --- flow coupling inverse (fully fused, 1 wave per 16-row tile) ------------
// All three weight matrices (104 KB f16) are staged once per block into LDS
// via async-to-LDS, then shared by 8 waves; fits easily in 320 KB/WGP.
__global__ __launch_bounds__(256) void flow_coupling_kernel(
    const float* __restrict__ yin, float* __restrict__ yout,
    const _Float16* __restrict__ W1h, const float* __restrict__ b1,
    const _Float16* __restrict__ W2h, const float* __restrict__ b2,
    const _Float16* __restrict__ W3h, const float* __restrict__ b3,
    float* __restrict__ logdet, int permute, int first) {
  __shared__ _Float16 sW1[64 * 256];   // 32 KB
  __shared__ _Float16 sW2[64 * 64];    //  8 KB
  __shared__ _Float16 sW3[512 * 64];   // 64 KB
  __shared__ float sH[8][16][Hc];      // 32 KB per-wave h staging

  int tid   = threadIdx.x;
  int lane  = tid & 31;
  int w     = tid >> 5;
  size_t row0 = ((size_t)blockIdx.x * 8 + w) * 16;
  int m     = lane & 15;
  int mhalf = (lane >> 4) * 8;

  // async-stage weights (16B chunks, ASYNCcnt-tracked, no VGPR transit)
  for (int c = tid; c < 64 * 256 / 8; c += 256) async_copy_b128(&sW1[c * 8], W1h + c * 8);
  for (int c = tid; c < 64 * 64 / 8;  c += 256) async_copy_b128(&sW2[c * 8], W2h + c * 8);
  for (int c = tid; c < 512 * 64 / 8; c += 256) async_copy_b128(&sW3[c * 8], W3h + c * 8);
  async_wait0();
  __syncthreads();

  v8f z8 = {};

  // h1 = relu(y1 @ W1^T + b1) : K = 256, N = 64
  v8f acc[4];
#pragma unroll
  for (int nt = 0; nt < 4; ++nt) acc[nt] = z8;
  for (int ks = 0; ks < 8; ++ks) {
    v16h a = load_frag_y(yin, row0, ks * 32, permute, lane);
#pragma unroll
    for (int nt = 0; nt < 4; ++nt) {
      v16h bw = load_frag_f16(sW1 + (size_t)(nt * 16) * 256 + ks * 32, 256, lane);
      acc[nt] = WMMA_F16(a, bw, acc[nt]);
    }
  }
#pragma unroll
  for (int nt = 0; nt < 4; ++nt)
#pragma unroll
    for (int j = 0; j < 8; ++j) {
      int n = nt * 16 + m;
      float v = acc[nt][j] + b1[n];
      sH[w][j + mhalf][n] = v > 0.f ? v : 0.f;
    }

  // h2 = relu(h1 @ W2^T + b2) : K = 64, N = 64
  v16h a0 = load_frag_f32(&sH[w][0][0], Hc, lane);
  v16h a1 = load_frag_f32(&sH[w][0][32], Hc, lane);
#pragma unroll
  for (int nt = 0; nt < 4; ++nt) {
    v8f a2 = z8;
    v16h bw0 = load_frag_f16(sW2 + (size_t)(nt * 16) * Hc, Hc, lane);
    v16h bw1 = load_frag_f16(sW2 + (size_t)(nt * 16) * Hc + 32, Hc, lane);
    a2 = WMMA_F16(a0, bw0, a2);
    a2 = WMMA_F16(a1, bw1, a2);
    acc[nt] = a2;
  }
#pragma unroll
  for (int nt = 0; nt < 4; ++nt)
#pragma unroll
    for (int j = 0; j < 8; ++j) {
      int n = nt * 16 + m;
      float v = acc[nt][j] + b2[n];
      sH[w][j + mhalf][n] = v > 0.f ? v : 0.f;
    }

  // st = h2 @ W3^T + b3 ; s-tile nt pairs with t-tile nt+16; fuse coupling
  v16h h0 = load_frag_f32(&sH[w][0][0], Hc, lane);
  v16h h1 = load_frag_f32(&sH[w][0][32], Hc, lane);
  float rowsum[8];
#pragma unroll
  for (int j = 0; j < 8; ++j) rowsum[j] = 0.f;

  for (int nt = 0; nt < 16; ++nt) {
    v8f accS = z8, accT = z8;
    v16h bs0 = load_frag_f16(sW3 + (size_t)(nt * 16) * Hc, Hc, lane);
    v16h bs1 = load_frag_f16(sW3 + (size_t)(nt * 16) * Hc + 32, Hc, lane);
    v16h bt0 = load_frag_f16(sW3 + (size_t)(HALFc + nt * 16) * Hc, Hc, lane);
    v16h bt1 = load_frag_f16(sW3 + (size_t)(HALFc + nt * 16) * Hc + 32, Hc, lane);
    accS = WMMA_F16(h0, bs0, accS);
    accS = WMMA_F16(h1, bs1, accS);
    accT = WMMA_F16(h0, bt0, accT);
    accT = WMMA_F16(h1, bt1, accT);
#pragma unroll
    for (int j = 0; j < 8; ++j) {
      int n = nt * 16 + m;
      size_t row = row0 + j + mhalf;
      float s = tanhf(accS[j] + b3[n]);
      float t = accT[j] + b3[HALFc + n];
      int c2 = HALFc + n;
      int c  = permute ? (Nc - 1 - c2) : c2;
      float y2 = yin[row * Nc + c];
      yout[row * Nc + c2] = (y2 - t) * __expf(-s);
      rowsum[j] -= s;
    }
  }

  // per-row logdet: reduce over the 16 lanes holding each row
#pragma unroll
  for (int j = 0; j < 8; ++j)
#pragma unroll
    for (int off = 1; off < 16; off <<= 1)
      rowsum[j] += __shfl_xor(rowsum[j], off, 32);
  if ((lane & 15) == 0) {
#pragma unroll
    for (int j = 0; j < 8; ++j) {
      size_t row = row0 + j + mhalf;
      if (first) logdet[row] = rowsum[j];
      else       logdet[row] += rowsum[j];
    }
  }

  // pass-through half: yout[:, :HALF] = (permuted) yin[:, :HALF]
  for (int idx = lane; idx < 16 * HALFc; idx += 32) {
    int r = idx >> 8, cc = idx & 255;
    size_t row = row0 + r;
    int c = permute ? (Nc - 1 - cc) : cc;
    yout[row * Nc + cc] = yin[row * Nc + c];
  }
}

// --- xp = relu(x @ proj_W^T + proj_b) ---------------------------------------
__global__ void proj_kernel(const float* __restrict__ x, const float* __restrict__ Wp,
                            const float* __restrict__ bp, float* __restrict__ xp) {
  int i = blockIdx.x * blockDim.x + threadIdx.x;
  if (i >= BT * PROJc) return;
  int row = i / PROJc, j = i % PROJc;
  const float* xr = x + (size_t)row * Kc;
  const float* wr = Wp + (size_t)j * Kc;
  float s = bp[j];
#pragma unroll
  for (int k = 0; k < Kc; ++k) s += xr[k] * wr[k];
  xp[i] = s > 0.f ? s : 0.f;
}

// --- LSTM layer scan (single block of 8 waves, WMMA gate GEMMs) -------------
// Gate weights (<=64 KB f16) are async-staged into LDS once and reused for all
// 512 timesteps.
__global__ __launch_bounds__(256) void lstm_kernel(
    const float* __restrict__ xin, const _Float16* __restrict__ Wc,
    const float* __restrict__ bih, const float* __restrict__ bhh,
    float* __restrict__ out, int in_dim, int kdim, int in_tb, int out_bt) {
  __shared__ _Float16 sW[256 * 128]; // packed [Wih|Whh|pad], 64 KB
  __shared__ _Float16 sA[32][128];   // [x_t | h_{t-1} | pad] as f16
  __shared__ float    sG[32][256];   // gate pre-activations
  __shared__ float    sHs[32][Hc];
  __shared__ float    sCs[32][Hc];
  int tid = threadIdx.x, lane = tid & 31, w = tid >> 5;

  for (int c = tid; c < 256 * kdim / 8; c += 256) async_copy_b128(&sW[c * 8], Wc + c * 8);
  for (int i = tid; i < 32 * Hc; i += 256) { sHs[i / Hc][i % Hc] = 0.f; sCs[i / Hc][i % Hc] = 0.f; }
  async_wait0();
  __syncthreads();

  int ksteps = kdim / 32;
  v8f z8 = {};
  for (int t = 0; t < Tc; ++t) {
    for (int i = tid; i < 32 * kdim; i += 256) {
      int b = i / kdim, k = i % kdim;
      float v;
      if (k < in_dim)
        v = in_tb ? xin[((size_t)t * Bc + b) * Hc + k]
                  : xin[((size_t)b * Tc + t) * in_dim + k];
      else if (k < in_dim + Hc)
        v = sHs[b][k - in_dim];
      else
        v = 0.f;
      sA[b][k] = (_Float16)v;
    }
    __syncthreads();

    // 2 M-tiles x 16 N-tiles across 8 waves (4 jobs each)
#pragma unroll
    for (int jj = 0; jj < 4; ++jj) {
      int job = w * 4 + jj;
      int mt = job >> 4, nt = job & 15;
      v8f accg = z8;
      for (int ks = 0; ks < ksteps; ++ks) {
        v16h a  = load_frag_f16(&sA[mt * 16][0] + ks * 32, 128, lane);
        v16h bw = load_frag_f16(sW + (size_t)(nt * 16) * kdim + ks * 32, kdim, lane);
        accg = WMMA_F16(a, bw, accg);
      }
      int mm = lane & 15, mh = (lane >> 4) * 8;
#pragma unroll
      for (int j = 0; j < 8; ++j)
        sG[mt * 16 + j + mh][nt * 16 + mm] = accg[j];
    }
    __syncthreads();

    for (int i = tid; i < 32 * Hc; i += 256) {
      int b = i / Hc, h = i % Hc;
      float gi = sG[b][h]          + bih[h]          + bhh[h];
      float gf = sG[b][Hc + h]     + bih[Hc + h]     + bhh[Hc + h];
      float gg = sG[b][2 * Hc + h] + bih[2 * Hc + h] + bhh[2 * Hc + h];
      float go = sG[b][3 * Hc + h] + bih[3 * Hc + h] + bhh[3 * Hc + h];
      float si = 1.f / (1.f + __expf(-gi));
      float sf = 1.f / (1.f + __expf(-gf));
      float so = 1.f / (1.f + __expf(-go));
      float c  = sf * sCs[b][h] + si * tanhf(gg);
      float hh = so * tanhf(c);
      sCs[b][h] = c;
      sHs[b][h] = hh;
      if (out_bt) out[((size_t)b * Tc + t) * Hc + h] = hh;
      else        out[((size_t)t * Bc + b) * Hc + h] = hh;
    }
    __syncthreads();
  }
}

// --- head GEMM + exp, writes (T,B,N,4) for coalesced Kalman reads -----------
__global__ __launch_bounds__(256) void head_kernel(
    const float* __restrict__ hseq,        // (B*T, H) rows = b*T+t
    const _Float16* __restrict__ Wh,       // (2048, 64) f16
    const float* __restrict__ bh,
    float* __restrict__ par) {             // (T, B, 2048)
  int lane = threadIdx.x & 31;
  int w    = threadIdx.x >> 5;
  size_t row0 = ((size_t)blockIdx.x * 8 + w) * 16;
  int m = lane & 15, mhalf = (lane >> 4) * 8;

  v16h a0 = load_frag_f32(hseq + row0 * Hc, Hc, lane);
  v16h a1 = load_frag_f32(hseq + row0 * Hc + 32, Hc, lane);
  v8f z8 = {};
  for (int nt = 0; nt < 128; ++nt) {
    v8f accg = z8;
    v16h bw0 = load_frag_f16(Wh + (size_t)(nt * 16) * Hc, Hc, lane);
    v16h bw1 = load_frag_f16(Wh + (size_t)(nt * 16) * Hc + 32, Hc, lane);
    accg = WMMA_F16(a0, bw0, accg);
    accg = WMMA_F16(a1, bw1, accg);
#pragma unroll
    for (int j = 0; j < 8; ++j) {
      int c = nt * 16 + m;
      size_t row = row0 + j + mhalf;
      int b = (int)(row / Tc), t = (int)(row % Tc);
      float v = accg[j] + bh[c];
      if ((c & 3) < 2) v = __expf(v);            // obs_var / proc_var
      par[((size_t)t * Bc + b) * 2048 + c] = v;
    }
  }
}

// --- D=2 Kalman scan, one thread per (b,n) ----------------------------------
__global__ __launch_bounds__(256) void kalman_kernel(
    const float* __restrict__ z,    // (B,T,N)
    const float* __restrict__ par,  // (T,B,2048) : n*4 + {ov,pv,a0,a1}
    float* __restrict__ outp) {
  int gid = blockIdx.x * blockDim.x + threadIdx.x;
  int b = gid / Nc, n = gid % Nc;
  float mu0 = 0.f, mu1 = 0.f, p00 = 1.f, p01 = 0.f, p10 = 0.f, p11 = 1.f;
  float ll = 0.f;
  const float LOG2PI = 1.8378770664093453f;
  for (int t = 0; t < Tc; ++t) {
    const float* pr = par + ((size_t)t * Bc + b) * 2048 + n * 4;
    if (t + 1 < Tc)
      __builtin_prefetch(par + ((size_t)(t + 1) * Bc + b) * 2048 + n * 4, 0, 1);
    float ov = pr[0], pv = pr[1], a0 = pr[2], a1 = pr[3];
    float zt = z[((size_t)b * Tc + t) * Nc + n];
    // predict: F = [[1,1],[0,1]]
    float m0 = mu0 + mu1, m1 = mu1;
    float q00 = p00 + p10 + p01 + p11 + pv;
    float q01 = p01 + p11;
    float q10 = p10 + p11;
    float q11 = p11 + pv;
    // update
    float pa0 = q00 * a0 + q01 * a1;
    float pa1 = q10 * a0 + q11 * a1;
    float S   = a0 * pa0 + a1 * pa1 + ov;
    float inv = 1.f / S;
    float k0 = pa0 * inv, k1 = pa1 * inv;
    float innov = zt - (a0 * m0 + a1 * m1);
    mu0 = m0 + k0 * innov;
    mu1 = m1 + k1 * innov;
    float ka00 = k0 * a0, ka01 = k0 * a1, ka10 = k1 * a0, ka11 = k1 * a1;
    p00 = q00 - (ka00 * q00 + ka01 * q10);
    p01 = q01 - (ka00 * q01 + ka01 * q11);
    p10 = q10 - (ka10 * q00 + ka11 * q10);
    p11 = q11 - (ka10 * q01 + ka11 * q11);
    ll += -0.5f * (innov * innov * inv + __logf(S) + LOG2PI);
  }
#pragma unroll
  for (int off = 1; off < 32; off <<= 1) ll += __shfl_xor(ll, off, 32);
  if ((threadIdx.x & 31) == 0) atomicAdd(outp, ll * (1.f / Bc));
}

__global__ void logdet_reduce_kernel(const float* __restrict__ ld, float* __restrict__ outp) {
  int gid = blockIdx.x * blockDim.x + threadIdx.x;
  float v = (gid < BT) ? ld[gid] : 0.f;
#pragma unroll
  for (int off = 1; off < 32; off <<= 1) v += __shfl_xor(v, off, 32);
  if ((threadIdx.x & 31) == 0) atomicAdd(outp, v * (1.f / Bc));
}

// ---------------------------------------------------------------------------
extern "C" void kernel_launch(void* const* d_in, const int* in_sizes, int n_in,
                              void* d_out, int out_size, void* d_ws, size_t ws_size,
                              hipStream_t stream) {
  (void)in_sizes; (void)n_in; (void)out_size; (void)ws_size;

  const float* y     = (const float*)d_in[0];
  const float* x     = (const float*)d_in[1];
  const float* projW = (const float*)d_in[26];
  const float* projB = (const float*)d_in[27];
  const float* Wih0  = (const float*)d_in[28];
  const float* bih0  = (const float*)d_in[29];
  const float* Whh0  = (const float*)d_in[30];
  const float* bhh0  = (const float*)d_in[31];
  const float* Wih1  = (const float*)d_in[32];
  const float* bih1  = (const float*)d_in[33];
  const float* Whh1  = (const float*)d_in[34];
  const float* bhh1  = (const float*)d_in[35];
  const float* headW = (const float*)d_in[36];
  const float* headB = (const float*)d_in[37];

  // workspace carve-up
  char* ws = (char*)d_ws;
  size_t off = 0;
  auto alloc = [&](size_t bytes) -> char* {
    char* p = ws + off;
    off += (bytes + 255) & ~(size_t)255;
    return p;
  };
  _Float16* fW1[4]; _Float16* fW2[4]; _Float16* fW3[4];
  for (int l = 0; l < 4; ++l) {
    fW1[l] = (_Float16*)alloc((size_t)64 * 256 * 2);
    fW2[l] = (_Float16*)alloc((size_t)64 * 64 * 2);
    fW3[l] = (_Float16*)alloc((size_t)512 * 64 * 2);
  }
  _Float16* lW0 = (_Float16*)alloc((size_t)256 * 96 * 2);
  _Float16* lW1 = (_Float16*)alloc((size_t)256 * 128 * 2);
  _Float16* hW  = (_Float16*)alloc((size_t)2048 * 64 * 2);
  float* zA   = (float*)alloc((size_t)BT * Nc * 4);
  float* zB   = (float*)alloc((size_t)BT * Nc * 4);
  float* ldet = (float*)alloc((size_t)BT * 4);
  float* xp   = (float*)alloc((size_t)BT * PROJc * 4);
  float* seq0 = (float*)alloc((size_t)BT * Hc * 4);  // (T,B,H)
  float* seq1 = (float*)alloc((size_t)BT * Hc * 4);  // (B,T,H)
  float* par  = (float*)alloc((size_t)BT * 2048 * 4);

  // weight prep
  for (int l = 0; l < 4; ++l) {
    const float* W1 = (const float*)d_in[2 + 6 * l + 0];
    const float* W2 = (const float*)d_in[2 + 6 * l + 2];
    const float* W3 = (const float*)d_in[2 + 6 * l + 4];
    cvt_f16_kernel<<<(64 * 256 + 255) / 256, 256, 0, stream>>>(W1, fW1[l], 64 * 256);
    cvt_f16_kernel<<<(64 * 64 + 255) / 256, 256, 0, stream>>>(W2, fW2[l], 64 * 64);
    cvt_f16_kernel<<<(512 * 64 + 255) / 256, 256, 0, stream>>>(W3, fW3[l], 512 * 64);
  }
  pack_lstm_kernel<<<(256 * 96 + 255) / 256, 256, 0, stream>>>(Wih0, Whh0, lW0, PROJc, 96);
  pack_lstm_kernel<<<(256 * 128 + 255) / 256, 256, 0, stream>>>(Wih1, Whh1, lW1, Hc, 128);
  cvt_f16_kernel<<<(2048 * 64 + 255) / 256, 256, 0, stream>>>(headW, hW, 2048 * 64);

  zero_kernel<<<1, 32, 0, stream>>>((float*)d_out);

  // flow inverse: reversed(flow) -> layers 3,2,1,0; permute before layers i>0
  float* outs[4] = {zA, zB, zA, zB};
  const float* fin = y;
  for (int i = 0; i < 4; ++i) {
    int l = 3 - i;
    const float* b1 = (const float*)d_in[2 + 6 * l + 1];
    const float* b2 = (const float*)d_in[2 + 6 * l + 3];
    const float* b3 = (const float*)d_in[2 + 6 * l + 5];
    flow_coupling_kernel<<<BT / 16 / 8, 256, 0, stream>>>(
        fin, outs[i], fW1[l], b1, fW2[l], b2, fW3[l], b3, ldet,
        (i > 0) ? 1 : 0, (i == 0) ? 1 : 0);
    fin = outs[i];
  }

  // param net
  proj_kernel<<<(BT * PROJc + 255) / 256, 256, 0, stream>>>(x, projW, projB, xp);
  lstm_kernel<<<1, 256, 0, stream>>>(xp,   lW0, bih0, bhh0, seq0, PROJc, 96, 0, 0);
  lstm_kernel<<<1, 256, 0, stream>>>(seq0, lW1, bih1, bhh1, seq1, Hc,   128, 1, 1);
  head_kernel<<<BT / 16 / 8, 256, 0, stream>>>(seq1, hW, headB, par);

  // likelihood
  kalman_kernel<<<(Bc * Nc) / 256, 256, 0, stream>>>(zB, par, (float*)d_out);
  logdet_reduce_kernel<<<(BT + 255) / 256, 256, 0, stream>>>(ldet, (float*)d_out);
}
